// DualSelfAttention_51831665328429
// MI455X (gfx1250) — matrix operand order repaired and checked by reading, not compiled, not run on hardware
//
#include <hip/hip_runtime.h>

// ---------------------------------------------------------------------------
// DualSelfAttention for MI455X (gfx1250, wave32, WMMA bf16)
// B=2, N=2048, H=1024, heads=16, d=64, SCALE=1/8
// Pipeline: fp32->bf16 convert (+weight transpose) -> QKV WMMA GEMM ->
//           flash-attention (WMMA + online softmax) -> out-proj WMMA GEMM.
// ---------------------------------------------------------------------------

typedef __attribute__((ext_vector_type(16))) __bf16 v16bf;
typedef __attribute__((ext_vector_type(8)))  __bf16 v8bf;
typedef __attribute__((ext_vector_type(8)))  float  v8f;

#define WMMA_BF16(A, B, C) \
  __builtin_amdgcn_wmma_f32_16x16x32_bf16(false, (A), false, (B), (short)0, (C), false, false)

// softmax scale folded with log2(e): exp(x*S) == exp2(x*S*log2e)
#define SCALE_LOG2E 0.1803368801111946f   // 0.125 * 1.4426950408889634

static __device__ __forceinline__ __bf16 f2bf(float f) {
  unsigned u = __builtin_bit_cast(unsigned, f);
  u += 0x7FFFu + ((u >> 16) & 1u);              // round-to-nearest-even
  unsigned short h = (unsigned short)(u >> 16);
  return __builtin_bit_cast(__bf16, h);
}

// A fragment: 16x32 bf16, row-major source with leading dim `ld` (elements).
// Layout: lane(l) -> row m=l&15; element e -> k = e + 16*(l>>4)
// => one contiguous 16-element (32B) chunk per lane.
static __device__ __forceinline__ v16bf load_a_frag(const __bf16* p, int ld, int lane) {
  const int m = lane & 15, hh = lane >> 4;
  return *(const v16bf*)(p + (size_t)m * ld + 16 * hh);   // k = 16h .. 16h+15
}

// B fragment: 32x16 bf16 where pT points at the TRANSPOSED operand
// (row n of pT = column n of B, contiguous over k; leading dim `ld`).
// Layout used here (contiguous k, same as A): lane -> column n=l&15; lanes 0-15 hold k=0..15, 16-31 hold k=16..31.
static __device__ __forceinline__ v16bf load_bt_frag(const __bf16* pT, int ld, int lane) {
  const int n = lane & 15, hh = lane >> 4;
  return *(const v16bf*)(pT + (size_t)n * ld + 16 * hh);   // one 32B load
}

// ---------------------------------------------------------------------------
// Stage 0: conversions
// ---------------------------------------------------------------------------
__global__ void cvt_bf16_kernel(const float* __restrict__ in, __bf16* __restrict__ out, int n) {
  int i = blockIdx.x * blockDim.x + threadIdx.x;
  if (i < n) out[i] = f2bf(in[i]);
}

// in[rows][cols] (fp32) -> out[cols][rows] (bf16)
__global__ void transpose_bf16_kernel(const float* __restrict__ in, __bf16* __restrict__ out,
                                      int rows, int cols) {
  int c = blockIdx.x * blockDim.x + threadIdx.x;
  int r = blockIdx.y;
  if (c < cols) out[(size_t)c * rows + r] = f2bf(in[(size_t)r * cols + c]);
}

// ---------------------------------------------------------------------------
// Stage 1: QKV projection.  M=4096 tokens, K=1024, Ncols=3072.
// part 0: Q = x @ Wqkv[:,0:1024]    -> Q  [B,h,N,d]
// part 1: K = x2 @ Wqkv[:,1024:2048]-> K  [B,h,N,d]
// part 2: V = x2 @ Wqkv[:,2048:3072]-> Vt [B,h,d,N]  (transposed for P@V)
// Block = 128x128 C tile, 8 waves (4x2), wave tile 32x64 (2x4 WMMA tiles).
// ---------------------------------------------------------------------------
__global__ __launch_bounds__(256) void qkv_gemm_kernel(
    const __bf16* __restrict__ xbf, const __bf16* __restrict__ x2bf,
    const __bf16* __restrict__ WT,   // [3072][1024]
    __bf16* __restrict__ Qo, __bf16* __restrict__ Ko, __bf16* __restrict__ Vt) {
  const int lane = threadIdx.x & 31;
  const int wid  = threadIdx.x >> 5;
  const int wm   = wid >> 1;                      // 0..3
  const int wn   = wid & 1;                       // 0..1
  const int row0 = blockIdx.x * 128 + wm * 32;    // token row
  const int col0 = blockIdx.y * 128 + wn * 64;    // qkv column (0..3071)
  const int part = col0 >> 10;                    // uniform per block
  const __bf16* A = (part == 0) ? xbf : x2bf;

  v8f acc[2][4] = {};
  for (int k0 = 0; k0 < 1024; k0 += 32) {
    v16bf a0 = load_a_frag(A + (size_t)row0 * 1024 + k0, 1024, lane);
    v16bf a1 = load_a_frag(A + (size_t)(row0 + 16) * 1024 + k0, 1024, lane);
#pragma unroll
    for (int t = 0; t < 4; ++t) {
      v16bf b = load_bt_frag(WT + (size_t)(col0 + 16 * t) * 1024 + k0, 1024, lane);
      acc[0][t] = WMMA_BF16(a0, b, acc[0][t]);
      acc[1][t] = WMMA_BF16(a1, b, acc[1][t]);
    }
  }

  const int nl = lane & 15, hh = lane >> 4;
#pragma unroll
  for (int i = 0; i < 2; ++i) {
#pragma unroll
    for (int t = 0; t < 4; ++t) {
#pragma unroll
      for (int r = 0; r < 8; ++r) {
        int row = row0 + i * 16 + r + 8 * hh;     // 0..4095
        int col = (col0 + t * 16 + nl) & 1023;    // column within part
        int b = row >> 11, tok = row & 2047;
        int head = col >> 6, d = col & 63;
        __bf16 v = f2bf(acc[i][t][r]);
        if (part == 0)      Qo[(((size_t)(b * 16 + head) * 2048 + tok) * 64) + d] = v;
        else if (part == 1) Ko[(((size_t)(b * 16 + head) * 2048 + tok) * 64) + d] = v;
        else                Vt[(((size_t)(b * 16 + head) * 64 + d) * 2048) + tok] = v;
      }
    }
  }
}

// ---------------------------------------------------------------------------
// Stage 2: flash attention.  Each wave owns 16 query rows; block = 8 waves =
// 128 rows of one (b,h).  kv streamed in chunks of 32; S and P@V via WMMA,
// P converted C-layout -> A-layout through wave-private LDS.
// ---------------------------------------------------------------------------
__global__ __launch_bounds__(256) void flash_attn_kernel(
    const __bf16* __restrict__ Q, const __bf16* __restrict__ K,
    const __bf16* __restrict__ Vt, __bf16* __restrict__ AttO) {
  __shared__ __bf16 pbuf[8][16 * 32];             // 8 KB: per-wave P staging
  const int lane = threadIdx.x & 31;
  const int wid  = threadIdx.x >> 5;
  const int bh   = blockIdx.y;                    // b*16+h
  const int q0   = blockIdx.x * 128 + wid * 16;
  const __bf16* Qh = Q  + (size_t)bh * 2048 * 64;
  const __bf16* Kh = K  + (size_t)bh * 2048 * 64;
  const __bf16* Vh = Vt + (size_t)bh * 64 * 2048;

  // Q fragments for this wave's 16 rows (d=64 -> two k-chunks), loaded once.
  v16bf qa0 = load_a_frag(Qh + (size_t)q0 * 64 +  0, 64, lane);
  v16bf qa1 = load_a_frag(Qh + (size_t)q0 * 64 + 32, 64, lane);

  v8f acc[4] = {};                                // 16x64 output accumulator
  float mrow[8], srow[8];
#pragma unroll
  for (int r = 0; r < 8; ++r) { mrow[r] = -__builtin_inff(); srow[r] = 0.f; }

  for (int kv = 0; kv < 2048; kv += 32) {
    // S = Q Kt : two 16x16 kv tiles, two d-chunks each
    v8f s0 = {}, s1 = {};
    s0 = WMMA_BF16(qa0, load_bt_frag(Kh + (size_t)kv * 64 +  0, 64, lane), s0);
    s0 = WMMA_BF16(qa1, load_bt_frag(Kh + (size_t)kv * 64 + 32, 64, lane), s0);
    s1 = WMMA_BF16(qa0, load_bt_frag(Kh + (size_t)(kv + 16) * 64 +  0, 64, lane), s1);
    s1 = WMMA_BF16(qa1, load_bt_frag(Kh + (size_t)(kv + 16) * 64 + 32, 64, lane), s1);

    // online softmax; C layout: VGPR r holds row (r + 8*(lane>>4)), col lane&15
#pragma unroll
    for (int r = 0; r < 8; ++r) {
      float v = fmaxf(s0[r], s1[r]);
      v = fmaxf(v, __shfl_xor(v, 1));
      v = fmaxf(v, __shfl_xor(v, 2));
      v = fmaxf(v, __shfl_xor(v, 4));
      v = fmaxf(v, __shfl_xor(v, 8));             // row max within 16-lane half
      float mnew  = fmaxf(mrow[r], v * SCALE_LOG2E);
      float alpha = exp2f(mrow[r] - mnew);
      mrow[r] = mnew;
      float p0 = exp2f(s0[r] * SCALE_LOG2E - mnew);
      float p1 = exp2f(s1[r] * SCALE_LOG2E - mnew);
      float ps = p0 + p1;
      ps += __shfl_xor(ps, 1);
      ps += __shfl_xor(ps, 2);
      ps += __shfl_xor(ps, 4);
      ps += __shfl_xor(ps, 8);
      srow[r] = srow[r] * alpha + ps;
#pragma unroll
      for (int t = 0; t < 4; ++t) acc[t][r] *= alpha;
      int row = r + 8 * (lane >> 4);
      pbuf[wid][row * 32 + (lane & 15)]      = f2bf(p0);
      pbuf[wid][row * 32 + 16 + (lane & 15)] = f2bf(p1);
    }
    __builtin_amdgcn_wave_barrier();              // keep DS store->load ordered

    // P @ V : A = P (16x32 from LDS), B = Vt rows (d-col contiguous over kv)
    v16bf pa = load_a_frag(&pbuf[wid][0], 32, lane);
#pragma unroll
    for (int t = 0; t < 4; ++t)
      acc[t] = WMMA_BF16(pa, load_bt_frag(Vh + (size_t)(16 * t) * 2048 + kv, 2048, lane), acc[t]);
  }

  // epilogue: normalize and scatter to AttO [B, N, H] bf16
  const int b = bh >> 4, head = bh & 15;
#pragma unroll
  for (int t = 0; t < 4; ++t) {
#pragma unroll
    for (int r = 0; r < 8; ++r) {
      int tok = q0 + r + 8 * (lane >> 4);
      int d   = 16 * t + (lane & 15);
      AttO[((size_t)(b * 2048 + tok)) * 1024 + head * 64 + d] = f2bf(acc[t][r] / srow[r]);
    }
  }
}

// ---------------------------------------------------------------------------
// Stage 3: output projection.  [4096,1024] bf16 @ WoutT + bout -> fp32 out.
// ---------------------------------------------------------------------------
__global__ __launch_bounds__(256) void out_gemm_kernel(
    const __bf16* __restrict__ A, const __bf16* __restrict__ WT,
    const float* __restrict__ bout, float* __restrict__ out) {
  const int lane = threadIdx.x & 31;
  const int wid  = threadIdx.x >> 5;
  const int wm   = wid >> 1;
  const int wn   = wid & 1;
  const int row0 = blockIdx.x * 128 + wm * 32;
  const int col0 = blockIdx.y * 128 + wn * 64;

  v8f acc[2][4] = {};
  for (int k0 = 0; k0 < 1024; k0 += 32) {
    v16bf a0 = load_a_frag(A + (size_t)row0 * 1024 + k0, 1024, lane);
    v16bf a1 = load_a_frag(A + (size_t)(row0 + 16) * 1024 + k0, 1024, lane);
#pragma unroll
    for (int t = 0; t < 4; ++t) {
      v16bf b = load_bt_frag(WT + (size_t)(col0 + 16 * t) * 1024 + k0, 1024, lane);
      acc[0][t] = WMMA_BF16(a0, b, acc[0][t]);
      acc[1][t] = WMMA_BF16(a1, b, acc[1][t]);
    }
  }

  const int nl = lane & 15, hh = lane >> 4;
#pragma unroll
  for (int i = 0; i < 2; ++i) {
#pragma unroll
    for (int t = 0; t < 4; ++t) {
#pragma unroll
      for (int r = 0; r < 8; ++r) {
        int row = row0 + i * 16 + r + 8 * hh;
        int col = col0 + t * 16 + nl;
        out[(size_t)row * 1024 + col] = acc[i][t][r] + bout[col];
      }
    }
  }
}

// ---------------------------------------------------------------------------
// Launcher.  Workspace layout (bf16 scratch, ~56 MB total):
//   0MB xbf | 8MB x2bf | 16MB WqkvT | 22MB WoutT | 24MB Q | 32MB K | 40MB Vt | 48MB AttO
// ---------------------------------------------------------------------------
extern "C" void kernel_launch(void* const* d_in, const int* in_sizes, int n_in,
                              void* d_out, int out_size, void* d_ws, size_t ws_size,
                              hipStream_t stream) {
  const float* x    = (const float*)d_in[0];
  const float* x2   = (const float*)d_in[1];
  const float* Wqkv = (const float*)d_in[2];
  const float* Wout = (const float*)d_in[3];
  const float* bout = (const float*)d_in[4];
  float* out = (float*)d_out;

  char* ws = (char*)d_ws;
  __bf16* xbf   = (__bf16*)(ws + ((size_t)0  << 20));
  __bf16* x2bf  = (__bf16*)(ws + ((size_t)8  << 20));
  __bf16* WqkvT = (__bf16*)(ws + ((size_t)16 << 20));
  __bf16* WoutT = (__bf16*)(ws + ((size_t)22 << 20));
  __bf16* Qb    = (__bf16*)(ws + ((size_t)24 << 20));
  __bf16* Kb    = (__bf16*)(ws + ((size_t)32 << 20));
  __bf16* Vtb   = (__bf16*)(ws + ((size_t)40 << 20));
  __bf16* AttO  = (__bf16*)(ws + ((size_t)48 << 20));

  const int NELT = 2 * 2048 * 1024;   // 4,194,304
  cvt_bf16_kernel<<<(NELT + 255) / 256, 256, 0, stream>>>(x, xbf, NELT);
  cvt_bf16_kernel<<<(NELT + 255) / 256, 256, 0, stream>>>(x2, x2bf, NELT);
  transpose_bf16_kernel<<<dim3(12, 1024), 256, 0, stream>>>(Wqkv, WqkvT, 1024, 3072);
  transpose_bf16_kernel<<<dim3(4, 1024),  256, 0, stream>>>(Wout, WoutT, 1024, 1024);

  qkv_gemm_kernel<<<dim3(32, 24), 256, 0, stream>>>(xbf, x2bf, WqkvT, Qb, Kb, Vtb);
  flash_attn_kernel<<<dim3(16, 32), 256, 0, stream>>>(Qb, Kb, Vtb, AttO);
  out_gemm_kernel<<<dim3(32, 8), 256, 0, stream>>>(AttO, WoutT, bout, out);
}
